// RelationalAttention_43611097924270
// MI455X (gfx1250) — matrix operand order repaired and checked by reading, not compiled, not run on hardware
//
#include <hip/hip_runtime.h>
#include <math.h>

#define N_NODES 4096
#define N_EDGES 8192
#define NODE_F  256
#define EDGE_F  128
#define D_OUT   128

typedef __attribute__((ext_vector_type(16))) __bf16 v16bf;
typedef __attribute__((ext_vector_type(8)))  float  v8f;

union Bf16Vec {
  v16bf v;
  uint4 u[2];
  unsigned short h[16];
};

// Hardware f32 -> bf16 conversion (backend selects native cvt on gfx1250).
__device__ __forceinline__ unsigned short f2bf(float f) {
  union { __bf16 b; unsigned short s; } o;
  o.b = (__bf16)f;
  return o.s;
}

__device__ __forceinline__ v8f wmma_bf16(v16bf a, v16bf b, v8f c) {
  // (neg_a, A, neg_b, B, c_mod, C, reuse_a, reuse_b)
  return __builtin_amdgcn_wmma_f32_16x16x32_bf16(false, a, false, b, (short)0, c,
                                                 false, false);
}

// Convert two contiguous 8-float chunks (32B-aligned) into a bf16 A/B fragment.
__device__ __forceinline__ v16bf cvt2x8(const float* p0, const float* p1) {
  const float4* a = (const float4*)p0;
  const float4* b = (const float4*)p1;
  float4 a0 = a[0], a1 = a[1], b0 = b[0], b1 = b[1];
  Bf16Vec r;
  r.h[0] = f2bf(a0.x); r.h[1] = f2bf(a0.y); r.h[2]  = f2bf(a0.z); r.h[3]  = f2bf(a0.w);
  r.h[4] = f2bf(a1.x); r.h[5] = f2bf(a1.y); r.h[6]  = f2bf(a1.z); r.h[7]  = f2bf(a1.w);
  r.h[8] = f2bf(b0.x); r.h[9] = f2bf(b0.y); r.h[10] = f2bf(b0.z); r.h[11] = f2bf(b0.w);
  r.h[12]= f2bf(b1.x); r.h[13]= f2bf(b1.y); r.h[14] = f2bf(b1.z); r.h[15] = f2bf(b1.w);
  return r.v;
}

// CDNA5 async memory->LDS (no VGPR data, tracked by ASYNCcnt).
#define ASYNC_LOAD_B128(dst_lds_off, src_ptr)                                  \
  asm volatile("global_load_async_to_lds_b128 %0, %1, off"                     \
               :: "v"(dst_lds_off), "v"(src_ptr) : "memory")
#define WAIT_ASYNC0() asm volatile("s_wait_asynccnt 0x0" ::: "memory")

// ---------------------------------------------------------------------------
// Phase 1: node projections  O[z][4096][128] = nodes @ W[z]^T + b[z], fp32 out
// block = 256 threads (8 waves), each wave: 16 rows x 128 cols, grid.z = q/k/v
// ---------------------------------------------------------------------------
__global__ void __launch_bounds__(256)
node_proj_kernel(const float* __restrict__ nodes,
                 const float* __restrict__ Wq, const float* __restrict__ bq,
                 const float* __restrict__ Wk, const float* __restrict__ bk,
                 const float* __restrict__ Wv, const float* __restrict__ bv,
                 float* __restrict__ oq, float* __restrict__ ok,
                 float* __restrict__ ov) {
  const float *W, *B; float* O;
  int z = blockIdx.z;
  if (z == 0)      { W = Wq; B = bq; O = oq; }
  else if (z == 1) { W = Wk; B = bk; O = ok; }
  else             { W = Wv; B = bv; O = ov; }

  const int wave = threadIdx.x >> 5, lane = threadIdx.x & 31;
  const int n = lane & 15, half = lane >> 4;
  const int row0 = blockIdx.x * 128 + wave * 16;
  const int arow_i = row0 + n;               // A-matrix row for this lane

  v8f acc[8];
  #pragma unroll
  for (int t = 0; t < 8; ++t) {
    float bb = B[t * 16 + n];                // bias depends only on column
    #pragma unroll
    for (int r = 0; r < 8; ++r) acc[t][r] = bb;
  }

  for (int k0 = 0; k0 < NODE_F; k0 += 32) {
    const float* ap = nodes + (size_t)arow_i * NODE_F + k0;
    v16bf a = cvt2x8(ap + half * 16, ap + half * 16 + 8);
    #pragma unroll
    for (int t = 0; t < 8; ++t) {
      const float* wp = W + (size_t)(t * 16 + n) * NODE_F + k0 + half * 16;
      v16bf b = cvt2x8(wp, wp + 8);          // B[k][n] = W[n][k], contiguous in k
      acc[t] = wmma_bf16(a, b, acc[t]);
    }
  }
  #pragma unroll
  for (int t = 0; t < 8; ++t)
    #pragma unroll
    for (int r = 0; r < 8; ++r)
      O[(size_t)(row0 + r + half * 8) * D_OUT + t * 16 + n] = acc[t][r];
}

// ---------------------------------------------------------------------------
// Phase 2: edge projections + gather/fuse.
//   q_ij = (ev @ Wq_e^T + bq_e) * nq[src]   -> bf16 [E][128]
//   k_ij = (ev @ Wk_e^T + bk_e) * nk[dst]   -> bf16 [E][128]
//   v_ij = (ev @ Wv_e^T + bv_e) * nv[dst]   -> bf16 TRANSPOSED [128][E]
// ---------------------------------------------------------------------------
__global__ void __launch_bounds__(256)
edge_proj_fuse_kernel(const float* __restrict__ ev, const int* __restrict__ eidx,
                      const float* __restrict__ Wq, const float* __restrict__ bq,
                      const float* __restrict__ Wk, const float* __restrict__ bk,
                      const float* __restrict__ Wv, const float* __restrict__ bv,
                      const float* __restrict__ nq, const float* __restrict__ nk,
                      const float* __restrict__ nv,
                      unsigned short* __restrict__ qij,
                      unsigned short* __restrict__ kij,
                      unsigned short* __restrict__ vT) {
  const float *W, *B, *NP; const int* idx;
  int z = blockIdx.z;
  if (z == 0)      { W = Wq; B = bq; NP = nq; idx = eidx; }            // src
  else if (z == 1) { W = Wk; B = bk; NP = nk; idx = eidx + N_EDGES; }  // dst
  else             { W = Wv; B = bv; NP = nv; idx = eidx + N_EDGES; }  // dst

  const int wave = threadIdx.x >> 5, lane = threadIdx.x & 31;
  const int n = lane & 15, half = lane >> 4;
  const int e0 = blockIdx.x * 128 + wave * 16;
  const int arow_i = e0 + n;

  v8f acc[8];
  #pragma unroll
  for (int t = 0; t < 8; ++t) {
    float bb = B[t * 16 + n];
    #pragma unroll
    for (int r = 0; r < 8; ++r) acc[t][r] = bb;
  }

  for (int k0 = 0; k0 < EDGE_F; k0 += 32) {
    const float* ap = ev + (size_t)arow_i * EDGE_F + k0;
    v16bf a = cvt2x8(ap + half * 16, ap + half * 16 + 8);
    #pragma unroll
    for (int t = 0; t < 8; ++t) {
      const float* wp = W + (size_t)(t * 16 + n) * EDGE_F + k0 + half * 16;
      v16bf b = cvt2x8(wp, wp + 8);
      acc[t] = wmma_bf16(a, b, acc[t]);
    }
  }

  #pragma unroll
  for (int r = 0; r < 8; ++r) {
    int e = e0 + r + half * 8;
    const float* nrow = NP + (size_t)idx[e] * D_OUT;
    #pragma unroll
    for (int t = 0; t < 8; ++t) {
      int col = t * 16 + n;
      unsigned short b = f2bf(acc[t][r] * nrow[col]);
      if (z == 0)      qij[(size_t)e * D_OUT + col] = b;
      else if (z == 1) kij[(size_t)e * D_OUT + col] = b;
      else             vT[(size_t)col * N_EDGES + e] = b;
    }
  }
}

// ---------------------------------------------------------------------------
// Phase 3: flash attention over edges.
// block = 128 threads (4 waves), 64 query rows/block, Bc = 64 key tile.
// Double-buffered K / V^T tiles staged with GLOBAL_LOAD_ASYNC_TO_LDS_B128
// (ASYNCcnt), so tile kt+1 streams into LDS while WMMAs chew on tile kt.
// Dynamic LDS: 2*16KB (K) + 2*16KB (V^T) + 8KB (P scratch) = 72KB.
// ---------------------------------------------------------------------------
#define KT_ELEMS (64 * 128)
#define SMEM_SHORTS (2 * KT_ELEMS + 2 * KT_ELEMS + 4 * 16 * 64)
#define SMEM_BYTES (SMEM_SHORTS * 2)

__global__ void __launch_bounds__(128)
edge_attention_kernel(const unsigned short* __restrict__ qij,
                      const unsigned short* __restrict__ kij,
                      const unsigned short* __restrict__ vT,
                      float* __restrict__ out_edges) {
  extern __shared__ unsigned short smem[];
  unsigned short* Kbuf = smem;                    // [2][64][128]
  unsigned short* Vbuf = smem + 2 * KT_ELEMS;     // [2][128][64]
  unsigned short* Psw  = smem + 4 * KT_ELEMS;     // [4][16][64]

  const int tid = threadIdx.x;
  const int wave = tid >> 5, lane = tid & 31;
  const int n = lane & 15, half = lane >> 4;
  const int q0 = blockIdx.x * 64 + wave * 16;
  const float sl = 0.08838834764831845f * 1.4426950408889634f; // scale * log2(e)

  // LDS byte offsets for the async DMA (flat shared addr low bits = LDS offset)
  const unsigned kbase = (unsigned)(size_t)Kbuf;
  const unsigned vbase = (unsigned)(size_t)Vbuf;

  // Preload Q A-fragments for this wave's 16 rows (K = 128 -> 4 fragments).
  Bf16Vec qa[4];
  const uint4* qrow = (const uint4*)(qij + (size_t)(q0 + n) * D_OUT);
  #pragma unroll
  for (int f = 0; f < 4; ++f) {
    qa[f].u[0] = qrow[f * 4 + half * 2];       // K = 32f + 16*half  .. +8
    qa[f].u[1] = qrow[f * 4 + half * 2 + 1];   // K = 32f + 16*half + 8 .. +8
  }

  v8f o[8];
  float mrow[8], lrow[8];
  #pragma unroll
  for (int dt = 0; dt < 8; ++dt)
    #pragma unroll
    for (int r = 0; r < 8; ++r) o[dt][r] = 0.0f;
  #pragma unroll
  for (int r = 0; r < 8; ++r) { mrow[r] = -3.0e38f; lrow[r] = 0.0f; }

  // Issue one tile's async loads into buffer `buf` (this thread's 8+8 chunks).
  auto issue_tile = [&](int kt, int buf) {
    const unsigned short* ks = kij + (size_t)kt * 64 * D_OUT;
    #pragma unroll
    for (int i = 0; i < 8; ++i) {
      int ci = tid + i * 128;                       // uint4 chunk index 0..1023
      unsigned kd = kbase + (unsigned)(buf * KT_ELEMS * 2) + (unsigned)(ci * 16);
      ASYNC_LOAD_B128(kd, (const void*)(ks + ci * 8));
      int d = ci >> 3, c = ci & 7;                  // V^T row d, col chunk c
      unsigned vd = vbase + (unsigned)(buf * KT_ELEMS * 2) + (unsigned)(ci * 16);
      ASYNC_LOAD_B128(vd, (const void*)(vT + (size_t)d * N_EDGES + kt * 64 + c * 8));
    }
  };

  issue_tile(0, 0);

  for (int kt = 0; kt < N_EDGES / 64; ++kt) {
    const int buf = kt & 1;
    WAIT_ASYNC0();            // this wave's async writes to LDS landed
    __syncthreads();          // all waves' tile data visible; prev compute done
    if (kt + 1 < N_EDGES / 64) issue_tile(kt + 1, buf ^ 1);

    const unsigned short* Kt = Kbuf + buf * KT_ELEMS;
    const unsigned short* Vt = Vbuf + buf * KT_ELEMS;

    // S = Q @ K^T  (16 x 64 per wave)
    v8f s[4];
    #pragma unroll
    for (int t = 0; t < 4; ++t)
      #pragma unroll
      for (int r = 0; r < 8; ++r) s[t][r] = 0.0f;
    #pragma unroll
    for (int f = 0; f < 4; ++f) {
      #pragma unroll
      for (int t = 0; t < 4; ++t) {
        Bf16Vec b;   // B[d][j] = K[j][d]: contiguous along d within row j
        const uint4* kp = (const uint4*)&Kt[(t * 16 + n) * 128 + f * 32 + half * 16];
        b.u[0] = kp[0]; b.u[1] = kp[1];
        s[t] = wmma_bf16(qa[f].v, b.v, s[t]);
      }
    }

    // Online softmax (rows live in 16-lane halves; xor{1,2,4,8} stays in-half).
    #pragma unroll
    for (int r = 0; r < 8; ++r) {
      float mx = fmaxf(fmaxf(s[0][r], s[1][r]), fmaxf(s[2][r], s[3][r]));
      mx = fmaxf(mx, __shfl_xor(mx, 1, 32));
      mx = fmaxf(mx, __shfl_xor(mx, 2, 32));
      mx = fmaxf(mx, __shfl_xor(mx, 4, 32));
      mx = fmaxf(mx, __shfl_xor(mx, 8, 32));
      float mn = fmaxf(mrow[r], mx);
      float alpha = exp2f((mrow[r] - mn) * sl);
      mrow[r] = mn;
      float rs = 0.0f;
      #pragma unroll
      for (int t = 0; t < 4; ++t) {
        float p = exp2f((s[t][r] - mn) * sl);
        s[t][r] = p;
        rs += p;
      }
      rs += __shfl_xor(rs, 1, 32);
      rs += __shfl_xor(rs, 2, 32);
      rs += __shfl_xor(rs, 4, 32);
      rs += __shfl_xor(rs, 8, 32);
      lrow[r] = lrow[r] * alpha + rs;
      #pragma unroll
      for (int dt = 0; dt < 8; ++dt) o[dt][r] *= alpha;
    }

    // C-layout -> A-layout via per-wave LDS round trip (wave LDS ops in-order).
    unsigned short* ps = Psw + wave * 16 * 64;
    #pragma unroll
    for (int r = 0; r < 8; ++r)
      #pragma unroll
      for (int t = 0; t < 4; ++t)
        ps[(r + half * 8) * 64 + t * 16 + n] = f2bf(s[t][r]);
    asm volatile("s_wait_dscnt 0x0" ::: "memory");

    // O += P @ V  (K = 64 -> 2 A-fragments)
    #pragma unroll
    for (int jf = 0; jf < 2; ++jf) {
      Bf16Vec pa;
      const unsigned short* pp = &ps[n * 64 + jf * 32 + half * 16];
      pa.u[0] = *(const uint4*)pp;
      pa.u[1] = *(const uint4*)(pp + 8);
      #pragma unroll
      for (int dt = 0; dt < 8; ++dt) {
        Bf16Vec b;  // B[j][d] = Vt[d][j]: contiguous along j
        const uint4* vp = (const uint4*)&Vt[(dt * 16 + n) * 64 + jf * 32 + half * 16];
        b.u[0] = vp[0]; b.u[1] = vp[1];
        o[dt] = wmma_bf16(pa.v, b.v, o[dt]);
      }
    }
  }

  #pragma unroll
  for (int r = 0; r < 8; ++r) {
    float inv = 1.0f / lrow[r];
    #pragma unroll
    for (int dt = 0; dt < 8; ++dt)
      out_edges[(size_t)(q0 + r + half * 8) * D_OUT + dt * 16 + n] = o[dt][r] * inv;
  }
}

// ---------------------------------------------------------------------------
// Phase 4: zero nodes, atomic scatter-sum of edge outputs by dst.
// ---------------------------------------------------------------------------
__global__ void __launch_bounds__(256)
zero_kernel(float* __restrict__ p, int count) {
  int i = blockIdx.x * 256 + threadIdx.x;
  if (i < count) p[i] = 0.0f;
}

__global__ void __launch_bounds__(256)
scatter_kernel(const float* __restrict__ out_edges, const int* __restrict__ eidx,
               float* __restrict__ out_nodes) {
  int i = blockIdx.x * 256 + threadIdx.x;   // over E*D
  int e = i >> 7, d = i & 127;
  atomicAdd(&out_nodes[(size_t)eidx[N_EDGES + e] * D_OUT + d], out_edges[i]);
}

extern "C" void kernel_launch(void* const* d_in, const int* in_sizes, int n_in,
                              void* d_out, int out_size, void* d_ws, size_t ws_size,
                              hipStream_t stream) {
  const float* nodes = (const float*)d_in[0];
  const int*   eidx  = (const int*)d_in[1];
  const float* ev    = (const float*)d_in[2];
  const float* wq_n = (const float*)d_in[3];  const float* bq_n = (const float*)d_in[4];
  const float* wq_e = (const float*)d_in[5];  const float* bq_e = (const float*)d_in[6];
  const float* wk_n = (const float*)d_in[7];  const float* bk_n = (const float*)d_in[8];
  const float* wk_e = (const float*)d_in[9];  const float* bk_e = (const float*)d_in[10];
  const float* wv_n = (const float*)d_in[11]; const float* bv_n = (const float*)d_in[12];
  const float* wv_e = (const float*)d_in[13]; const float* bv_e = (const float*)d_in[14];

  float* out_nodes = (float*)d_out;
  float* out_edges = (float*)d_out + (size_t)N_NODES * D_OUT;

  // Workspace layout (12 MB total):
  float* nq = (float*)d_ws;
  float* nk = nq + (size_t)N_NODES * D_OUT;
  float* nv = nk + (size_t)N_NODES * D_OUT;
  unsigned short* qij = (unsigned short*)(nv + (size_t)N_NODES * D_OUT);
  unsigned short* kij = qij + (size_t)N_EDGES * D_OUT;
  unsigned short* vT  = kij + (size_t)N_EDGES * D_OUT;

  node_proj_kernel<<<dim3(N_NODES / 128, 1, 3), 256, 0, stream>>>(
      nodes, wq_n, bq_n, wk_n, bk_n, wv_n, bv_n, nq, nk, nv);
  edge_proj_fuse_kernel<<<dim3(N_EDGES / 128, 1, 3), 256, 0, stream>>>(
      ev, eidx, wq_e, bq_e, wk_e, bk_e, wv_e, bv_e, nq, nk, nv, qij, kij, vT);
  edge_attention_kernel<<<dim3(N_EDGES / 64), 128, SMEM_BYTES, stream>>>(
      qij, kij, vT, out_edges);
  zero_kernel<<<dim3((N_NODES * D_OUT + 255) / 256), 256, 0, stream>>>(
      out_nodes, N_NODES * D_OUT);
  scatter_kernel<<<dim3((N_EDGES * D_OUT) / 256), 256, 0, stream>>>(
      out_edges, eidx, out_nodes);
}
